// dMoA_2216203124860
// MI455X (gfx1250) — matrix operand order repaired and checked by reading, not compiled, not run on hardware
//
#include <hip/hip_runtime.h>
#include <hip/hip_bf16.h>
#include <cstddef>

// ---------------------------------------------------------------------------
// dMoA forward for MI455X (gfx1250, wave32, WMMA 16x16x32 f16)
// Grouped GEMMs with register-resident A: each wave owns 16 rows whose
// A-fragments (K=1024) live in 256 VGPRs (gfx1250 VGPR-MSB addressing),
// so the steady-state loop is clause'd global_load_b128 of B + back-to-back
// v_wmma (no LDS, no hazard nops). 8 waves/block share identical B-fragment
// addresses (M-tile=128) -> weight stream read once per tile via WGP$/L2.
// ---------------------------------------------------------------------------
static constexpr int SL = 2048, BS = 4, H = 1024, F = 1024, E = 8, TOPK = 2;
static constexpr int T = SL * BS;                  // 8192 tokens
static constexpr int NENT = T * TOPK;              // 16384 routed entries
static constexpr int TILE_M = 128;                 // 8 waves x 16 rows
static constexpr int MAX_TILES = NENT / TILE_M + E;   // 136 worst-case tiles
static constexpr int NBLK = 4;                     // N split across gridDim.y
static constexpr int NCOLS = F / NBLK;             // 256 cols per block

typedef __attribute__((ext_vector_type(16))) _Float16 v16h;
typedef __attribute__((ext_vector_type(8)))  float    v8f;

union F16Frag { uint4 q[2]; v16h v; };             // 32 bytes = 8 VGPRs

// ---- workspace layout (bytes) ----
static constexpr size_t WS_CTRL   = 0;             // ints: counts[8] cursor[8] nt offsets[8]
static constexpr int CTRL_CNT = 0, CTRL_CUR = 8, CTRL_NT = 16, CTRL_OFF = 24;
static constexpr size_t WS_PART   = 512;           // float[32*8] router prob partials
static constexpr size_t WS_TILES  = 1536;          // int[MAX_TILES*3]
static constexpr size_t WS_TOPE   = 16384;         // int[T*2]
static constexpr size_t WS_TOPW   = 81920;         // float[T*2]
static constexpr size_t WS_PERM   = 147456;        // int[NENT]
static constexpr size_t WS_ENTW   = 212992;        // float[NENT]
static constexpr size_t WS_XH     = 1u << 20;                       // f16[T*H]     16 MB
static constexpr size_t WS_W1T    = WS_XH  + (size_t)T * H * 2;     // f16[E*F*H]   16 MB
static constexpr size_t WS_W2T    = WS_W1T + (size_t)E * F * H * 2; // f16[E*H*F]   16 MB
static constexpr size_t WS_HBUF   = WS_W2T + (size_t)E * H * F * 2; // f16[MAX_TILES*128*F] ~35.7 MB

__device__ __forceinline__ int imin(int a, int b) { return a < b ? a : b; }

// ---------------------------------------------------------------------------
// 0) zero output accumulator + control region
// ---------------------------------------------------------------------------
__global__ void zero_kernel(float* __restrict__ out, size_t n, int* __restrict__ ctrl) {
    size_t stride = (size_t)gridDim.x * blockDim.x;
    for (size_t i = (size_t)blockIdx.x * blockDim.x + threadIdx.x; i < n; i += stride)
        out[i] = 0.0f;
    size_t j = (size_t)blockIdx.x * blockDim.x + threadIdx.x;
    if (j < 384) ctrl[j] = 0;   // counts/cursors/nt/offsets + partials
}

// ---------------------------------------------------------------------------
// 1) precision/layout conversion kernels (fp32 -> f16, weights transposed)
// ---------------------------------------------------------------------------
__global__ void cvt_x_kernel(const float* __restrict__ x, _Float16* __restrict__ xh, size_t n) {
    size_t stride = (size_t)gridDim.x * blockDim.x;
    for (size_t i = (size_t)blockIdx.x * blockDim.x + threadIdx.x; i < n; i += stride)
        xh[i] = (_Float16)x[i];
}
__global__ void cvt_w1_kernel(const float* __restrict__ w1, _Float16* __restrict__ w1t) {
    size_t n = (size_t)E * H * F, stride = (size_t)gridDim.x * blockDim.x;
    for (size_t i = (size_t)blockIdx.x * blockDim.x + threadIdx.x; i < n; i += stride) {
        int e = (int)(i / ((size_t)H * F));
        int rem = (int)(i - (size_t)e * H * F);
        int h = rem / F, f = rem % F;
        w1t[((size_t)e * F + f) * H + h] = (_Float16)w1[i];   // [e][f][h]
    }
}
__global__ void cvt_w2_kernel(const float* __restrict__ w2, _Float16* __restrict__ w2t) {
    size_t n = (size_t)E * F * H, stride = (size_t)gridDim.x * blockDim.x;
    for (size_t i = (size_t)blockIdx.x * blockDim.x + threadIdx.x; i < n; i += stride) {
        int e = (int)(i / ((size_t)F * H));
        int rem = (int)(i - (size_t)e * F * H);
        int f = rem / H, hh = rem % H;
        w2t[((size_t)e * H + hh) * F + f] = (_Float16)w2[i];  // [e][h'][f]
    }
}

// ---------------------------------------------------------------------------
// 2) router: logits -> softmax -> top2 ; counts (int atomics) ; prob partials
// ---------------------------------------------------------------------------
__global__ __launch_bounds__(256) void router_kernel(
    const float* __restrict__ x, const float* __restrict__ wr,
    int* __restrict__ top_e, float* __restrict__ top_w,
    int* __restrict__ ctrl, float* __restrict__ partials) {
    __shared__ float swr[H * E];   // 32 KB router weights
    __shared__ float red[256];
    int tid = threadIdx.x;
    for (int i = tid; i < H * E; i += 256) swr[i] = wr[i];
    __syncthreads();

    int t = blockIdx.x * 256 + tid;   // grid sized exactly T/256
    float acc[E] = {0, 0, 0, 0, 0, 0, 0, 0};
    const float4* xr = (const float4*)(x + (size_t)t * H);
    for (int h4 = 0; h4 < H / 4; ++h4) {
        float4 xv = xr[h4];
        int h = h4 * 4;
#pragma unroll
        for (int e = 0; e < E; ++e) {
            acc[e] += xv.x * swr[(h + 0) * E + e] + xv.y * swr[(h + 1) * E + e]
                    + xv.z * swr[(h + 2) * E + e] + xv.w * swr[(h + 3) * E + e];
        }
    }
    float mx = acc[0];
#pragma unroll
    for (int e = 1; e < E; ++e) mx = fmaxf(mx, acc[e]);
    float p[E], s = 0.0f;
#pragma unroll
    for (int e = 0; e < E; ++e) { p[e] = expf(acc[e] - mx); s += p[e]; }
    float inv = 1.0f / s;
#pragma unroll
    for (int e = 0; e < E; ++e) p[e] *= inv;

    int e0 = 0;
#pragma unroll
    for (int e = 1; e < E; ++e) if (p[e] > p[e0]) e0 = e;
    int e1 = (e0 == 0) ? 1 : 0;
#pragma unroll
    for (int e = 0; e < E; ++e) if (e != e0 && p[e] > p[e1]) e1 = e;

    top_e[t * 2 + 0] = e0;  top_w[t * 2 + 0] = p[e0];
    top_e[t * 2 + 1] = e1;  top_w[t * 2 + 1] = p[e1];
    atomicAdd(&ctrl[CTRL_CNT + e0], 1);
    atomicAdd(&ctrl[CTRL_CNT + e1], 1);

    // deterministic per-block reduction of softmax probs (for the aux loss)
    for (int e = 0; e < E; ++e) {
        red[tid] = p[e];
        __syncthreads();
        for (int st = 128; st > 0; st >>= 1) {
            if (tid < st) red[tid] += red[tid + st];
            __syncthreads();
        }
        if (tid == 0) partials[blockIdx.x * E + e] = red[0];
        __syncthreads();
    }
}

// ---------------------------------------------------------------------------
// 3) plan: offsets + 128-row tile descriptors + load-balancing loss
// ---------------------------------------------------------------------------
__global__ void plan_kernel(int* __restrict__ ctrl, int* __restrict__ tiles,
                            const float* __restrict__ partials, float* __restrict__ loss_out) {
    if (threadIdx.x != 0 || blockIdx.x != 0) return;
    int offs[E], off = 0;
    for (int e = 0; e < E; ++e) { offs[e] = off; ctrl[CTRL_OFF + e] = off; off += ctrl[CTRL_CNT + e]; }
    int nt = 0;
    for (int e = 0; e < E; ++e) {
        int end = offs[e] + ctrl[CTRL_CNT + e];
        for (int r = offs[e]; r < end; r += TILE_M) {
            tiles[nt * 3 + 0] = e;
            tiles[nt * 3 + 1] = r;
            tiles[nt * 3 + 2] = imin(TILE_M, end - r);
            ++nt;
        }
    }
    ctrl[CTRL_NT] = nt;
    float loss = 0.0f;
    for (int e = 0; e < E; ++e) {
        float ps = 0.0f;
        for (int b = 0; b < T / 256; ++b) ps += partials[b * E + e];
        float fe = (float)ctrl[CTRL_CNT + e] / (float)(T * TOPK);
        loss += fe * (ps / (float)T);
    }
    loss_out[0] = loss * (float)E;
}

// ---------------------------------------------------------------------------
// 4) assign: dropless scatter of tokens into per-expert entry lists
// ---------------------------------------------------------------------------
__global__ void assign_kernel(const int* __restrict__ top_e, const float* __restrict__ top_w,
                              int* __restrict__ ctrl, int* __restrict__ perm,
                              float* __restrict__ entw) {
    int t = blockIdx.x * blockDim.x + threadIdx.x;
    if (t >= T) return;
#pragma unroll
    for (int k = 0; k < TOPK; ++k) {
        int e = top_e[t * 2 + k];
        int pos = atomicAdd(&ctrl[CTRL_CUR + e], 1);
        int slot = ctrl[CTRL_OFF + e] + pos;
        perm[slot] = t;
        entw[slot] = top_w[t * 2 + k];
    }
}

// ---------------------------------------------------------------------------
// A-fragment register file: per-wave 16 rows x K=1024 -> 32 F16Frags (256 VGPR)
// Layout: lane holds M=lane&15; element j -> K = k0 + (lane/16)*16 + j
// => two 16B chunks at k0+half*16 and k0+half*16+8 (same k order as B).
// B 32x16: lane holds N = base+(lane&15); 16 contiguous f16 at K = k0+half*16.
// C 16x16: elem r -> (M=(lane/16)*8+r, N=lane&15).
// ---------------------------------------------------------------------------

// 5) map GEMM: per 128-entry tile, Htile = Xg(128xH) @ w1[e](HxF)
__global__ __launch_bounds__(256) void moe_map_kernel(
    const _Float16* __restrict__ xh, const _Float16* __restrict__ w1t,
    const int* __restrict__ ctrl, const int* __restrict__ tiles,
    const int* __restrict__ perm, _Float16* __restrict__ hbuf) {
    int b = blockIdx.x;
    if (b >= ctrl[CTRL_NT]) return;
    int e = tiles[b * 3 + 0], row0 = tiles[b * 3 + 1], valid = tiles[b * 3 + 2];
    int tid = threadIdx.x;
    int lane = tid & 31, wave = tid >> 5;
    int half = lane >> 4, l16 = lane & 15;

    // A rows: wave owns rows [wave*16, wave*16+16); lane's row = wave*16+l16
    int mA = wave * 16 + l16;
    int tok = (mA < valid) ? perm[row0 + mA] : 0;
    const _Float16* arow = xh + (size_t)tok * H;

    F16Frag areg[32];                                  // 256 VGPRs, K-resident
#pragma unroll
    for (int kk = 0; kk < 32; ++kk) {
        areg[kk].q[0] = *(const uint4*)(arow + kk * 32 + half * 16);
        areg[kk].q[1] = *(const uint4*)(arow + kk * 32 + half * 16 + 8);
    }

    const _Float16* w1te = w1t + (size_t)e * F * H;
    int nbase = blockIdx.y * NCOLS;
    for (int fi = 0; fi < NCOLS / 16; ++fi) {          // 16 N-slices of 16
        int fc = nbase + fi * 16;
        const _Float16* brow = w1te + (size_t)(fc + l16) * H;  // same addr in all 8 waves
        v8f acc = {};
#pragma unroll
        for (int kk = 0; kk < 32; ++kk) {
            F16Frag bf;
            const uint4* bp = (const uint4*)(brow + kk * 32 + half * 16);
            bf.q[0] = bp[0];  bf.q[1] = bp[1];
            acc = __builtin_amdgcn_wmma_f32_16x16x32_f16(
                false, areg[kk].v, false, bf.v, (short)0, acc, false, false);
        }
#pragma unroll
        for (int r = 0; r < 8; ++r) {                  // C layout scatter
            int m = wave * 16 + half * 8 + r;
            hbuf[((size_t)b * TILE_M + m) * F + fc + l16] = (_Float16)acc[r];
        }
    }
}

// 6) reduce GEMM: Out_tile = Htile(128xF) @ w2[e](FxH), weighted scatter-add
__global__ __launch_bounds__(256) void moe_reduce_kernel(
    const _Float16* __restrict__ hbuf, const _Float16* __restrict__ w2t,
    const int* __restrict__ ctrl, const int* __restrict__ tiles,
    const int* __restrict__ perm, const float* __restrict__ entw,
    float* __restrict__ out) {
    __shared__ int   s_tok[TILE_M];
    __shared__ float s_w[TILE_M];
    int b = blockIdx.x;
    if (b >= ctrl[CTRL_NT]) return;
    int e = tiles[b * 3 + 0], row0 = tiles[b * 3 + 1], valid = tiles[b * 3 + 2];
    int tid = threadIdx.x;
    if (tid < TILE_M) {
        bool v = tid < valid;
        s_tok[tid] = v ? perm[row0 + tid] : 0;
        s_w[tid]   = v ? entw[row0 + tid] : 0.0f;
    }
    __syncthreads();

    int lane = tid & 31, wave = tid >> 5;
    int half = lane >> 4, l16 = lane & 15;

    int mA = wave * 16 + l16;
    const _Float16* arow = hbuf + ((size_t)b * TILE_M + mA) * F;

    F16Frag areg[32];                                  // 256 VGPRs, K-resident
#pragma unroll
    for (int kk = 0; kk < 32; ++kk) {
        areg[kk].q[0] = *(const uint4*)(arow + kk * 32 + half * 16);
        areg[kk].q[1] = *(const uint4*)(arow + kk * 32 + half * 16 + 8);
    }

    const _Float16* w2te = w2t + (size_t)e * H * F;
    int nbase = blockIdx.y * NCOLS;
    for (int hi = 0; hi < NCOLS / 16; ++hi) {
        int hc = nbase + hi * 16;
        const _Float16* brow = w2te + (size_t)(hc + l16) * F;  // shared across waves
        v8f acc = {};
#pragma unroll
        for (int kk = 0; kk < 32; ++kk) {
            F16Frag bf;
            const uint4* bp = (const uint4*)(brow + kk * 32 + half * 16);
            bf.q[0] = bp[0];  bf.q[1] = bp[1];
            acc = __builtin_amdgcn_wmma_f32_16x16x32_f16(
                false, areg[kk].v, false, bf.v, (short)0, acc, false, false);
        }
#pragma unroll
        for (int r = 0; r < 8; ++r) {
            int m = wave * 16 + half * 8 + r;
            if (m < valid)
                atomicAdd(&out[(size_t)s_tok[m] * H + hc + l16], s_w[m] * acc[r]);
        }
    }
}

// ---------------------------------------------------------------------------
extern "C" void kernel_launch(void* const* d_in, const int* in_sizes, int n_in,
                              void* d_out, int out_size, void* d_ws, size_t ws_size,
                              hipStream_t stream) {
    (void)in_sizes; (void)n_in; (void)out_size; (void)ws_size;
    const float* x  = (const float*)d_in[0];
    const float* wr = (const float*)d_in[1];
    const float* w1 = (const float*)d_in[2];
    const float* w2 = (const float*)d_in[3];
    float* out = (float*)d_out;

    char* ws = (char*)d_ws;
    int*      ctrl  = (int*)(ws + WS_CTRL);
    float*    part  = (float*)(ws + WS_PART);
    int*      tiles = (int*)(ws + WS_TILES);
    int*      tope  = (int*)(ws + WS_TOPE);
    float*    topw  = (float*)(ws + WS_TOPW);
    int*      perm  = (int*)(ws + WS_PERM);
    float*    entw  = (float*)(ws + WS_ENTW);
    _Float16* xh    = (_Float16*)(ws + WS_XH);
    _Float16* w1t   = (_Float16*)(ws + WS_W1T);
    _Float16* w2t   = (_Float16*)(ws + WS_W2T);
    _Float16* hbuf  = (_Float16*)(ws + WS_HBUF);

    zero_kernel<<<2048, 256, 0, stream>>>(out, (size_t)T * H, ctrl);
    cvt_x_kernel<<<2048, 256, 0, stream>>>(x, xh, (size_t)T * H);
    cvt_w1_kernel<<<4096, 256, 0, stream>>>(w1, w1t);
    cvt_w2_kernel<<<4096, 256, 0, stream>>>(w2, w2t);
    router_kernel<<<T / 256, 256, 0, stream>>>(x, wr, tope, topw, ctrl, part);
    plan_kernel<<<1, 32, 0, stream>>>(ctrl, tiles, part, out + (size_t)T * H);
    assign_kernel<<<T / 256, 256, 0, stream>>>(tope, topw, ctrl, perm, entw);
    dim3 ggrid(MAX_TILES, NBLK);
    moe_map_kernel<<<ggrid, 256, 0, stream>>>(xh, w1t, ctrl, tiles, perm, hbuf);
    moe_reduce_kernel<<<ggrid, 256, 0, stream>>>(hbuf, w2t, ctrl, tiles, perm, entw, out);
}
